// Transformer_4329327035207
// MI455X (gfx1250) — matrix operand order repaired and checked by reading, not compiled, not run on hardware
//
#include <hip/hip_runtime.h>
#include <hip/hip_bf16.h>

// MI455X / gfx1250: wave32, WMMA 16x16x32 bf16 -> f32 accum.
// Block-tiled GEMM: 128x64 C per 256-thread block (8 waves x 32x32),
// K staged in 32-wide chunks through double-buffered LDS via the
// Tensor Data Mover (TENSOR_LOAD_TO_LDS + s_wait_tensorcnt), fallback
// to cooperative global->LDS copies if the TDM builtin is absent.

typedef __bf16 bf16;
typedef __bf16 v16bf __attribute__((ext_vector_type(16)));
typedef __bf16 v8bf  __attribute__((ext_vector_type(8)));
typedef float  v8f   __attribute__((ext_vector_type(8)));
typedef unsigned int u32x4 __attribute__((ext_vector_type(4)));
typedef int  i32x4 __attribute__((ext_vector_type(4)));
typedef int  i32x8 __attribute__((ext_vector_type(8)));

#define DMODEL 1024
#define BATCH  4
#define SEQ    2048
#define MTOK   (BATCH * SEQ)   // 8192

#if defined(__has_builtin)
#  if __has_builtin(__builtin_amdgcn_tensor_load_to_lds) && \
      __has_builtin(__builtin_amdgcn_s_wait_tensorcnt)
#    define USE_TDM 1
#  else
#    define USE_TDM 0
#  endif
#else
#  define USE_TDM 0
#endif

// Block tile geometry
#define BM 128
#define BN 64
#define BK 32
#define A_ELEMS (BM * BK)           // 4096 bf16 = 8 KB
#define B_ELEMS (BN * BK)           // 2048 bf16 = 4 KB
#define BUF_ELEMS (A_ELEMS + B_ELEMS)

// ---- WMMA fragment loads (ISA 7.12.2 layouts, wave32) ----
__device__ __forceinline__ v16bf load_frag_a(const bf16* Abase, int ld, int mofs,
                                             int k0, int lane) {
  int m = (lane & 15) + mofs;
  int h = lane >> 4;
  const bf16* p = Abase + (size_t)m * ld + k0 + 8 * h;
  v8bf lo = *(const v8bf*)p;
  v8bf hi = *(const v8bf*)(p + 16);
  v16bf f;
#pragma unroll
  for (int i = 0; i < 8; ++i) { f[i] = lo[i]; f[8 + i] = hi[i]; }
  return f;
}

__device__ __forceinline__ v16bf load_frag_b(const bf16* Btbase, int ld, int nofs,
                                             int k0, int lane) {
  int n = (lane & 15) + nofs;
  int h = lane >> 4;
  const bf16* p = Btbase + (size_t)n * ld + k0 + 8 * h;
  v8bf lo = *(const v8bf*)p;
  v8bf hi = *(const v8bf*)(p + 16);
  v16bf f;
#pragma unroll
  for (int i = 0; i < 8; ++i) { f[i] = lo[i]; f[8 + i] = hi[i]; }
  return f;
}

__device__ __forceinline__ void store_tile(void* C, int mode, int ldC,
                                           int m0, int n0, int lane,
                                           v8f acc, float scale) {
  int n  = n0 + (lane & 15);
  int mb = m0 + (lane >> 4) * 8;
  if (mode == 0) {
    bf16* c = (bf16*)C;
#pragma unroll
    for (int r = 0; r < 8; ++r) c[(size_t)(mb + r) * ldC + n] = (bf16)(acc[r] * scale);
  } else if (mode == 1) {
    bf16* c = (bf16*)C;
    v8bf pk;
#pragma unroll
    for (int r = 0; r < 8; ++r) pk[r] = (bf16)(acc[r] * scale);
    *(v8bf*)(c + (size_t)n * ldC + mb) = pk;
  } else {
    float* c = (float*)C;
#pragma unroll
    for (int r = 0; r < 8; ++r) c[(size_t)(mb + r) * ldC + n] = acc[r] * scale;
  }
}

#if USE_TDM
// Issue a 2-D TDM load: tile_d1 rows x tile_d0 elems (2-byte) from a tensor
// with row stride 'stride_elems', into LDS at byte offset lds_off.
// D# layout per CDNA5 ISA 8.3 / 8.4 (count=1, type=2 image, data_size=2B).
__device__ __forceinline__ void tdm_load_2d(const void* gptr, unsigned lds_off,
                                            unsigned tile_d0, unsigned tile_d1,
                                            unsigned long long stride_elems) {
  unsigned long long ga = (unsigned long long)gptr;
  const unsigned TD = 0x40000000u;  // huge tensor dims: tiles never clip
  u32x4 g0;
  g0[0] = 1u;                                   // count=1, user descriptor
  g0[1] = lds_off;                              // lds_addr (bytes)
  g0[2] = (unsigned)(ga & 0xffffffffu);         // global_addr[31:0]
  g0[3] = (unsigned)((ga >> 32) & 0x1ffffffu)   // global_addr[56:32]
          | (2u << 30);                         // type = 2
  i32x8 g1;
  g1[0] = (int)(1u << 16);                      // data_size = 1 (2 bytes)
  g1[1] = (int)((TD & 0xffffu) << 16);          // tensor_dim0[15:0]
  g1[2] = (int)((TD >> 16) | ((TD & 0xffffu) << 16));   // td0 hi | td1 lo
  g1[3] = (int)((TD >> 16) | (tile_d0 << 16));  // td1 hi | tile_dim0
  g1[4] = (int)(tile_d1 & 0xffffu);             // tile_dim1 | tile_dim2=0
  g1[5] = (int)(unsigned)(stride_elems & 0xffffffffu);  // dim0_stride lo32
  g1[6] = (int)(unsigned)((stride_elems >> 32) & 0xffffu); // stride hi16 | d1s lo16=0
  g1[7] = 0;
  i32x4 gz = {0, 0, 0, 0};
#if __clang_major__ >= 23
  i32x8 gz8 = {0, 0, 0, 0, 0, 0, 0, 0};
  __builtin_amdgcn_tensor_load_to_lds(g0, g1, gz, gz, gz8, 0);
#else
  __builtin_amdgcn_tensor_load_to_lds(g0, g1, gz, gz, 0);
#endif
}
#endif

// Stage one K-chunk (A: BMxBK, B: BNxBK) into LDS buffer 'buf'.
__device__ __forceinline__ void stage_chunk(const bf16* Ab, const bf16* Bb,
                                            int ldA, int ldB, int k0,
                                            bf16* smem, unsigned smem_base,
                                            int buf, int tid) {
  bf16* sA = smem + buf * BUF_ELEMS;
  bf16* sB = sA + A_ELEMS;
#if USE_TDM
  (void)tid;
  unsigned lofs = smem_base + (unsigned)(buf * BUF_ELEMS) * 2u;
  tdm_load_2d(Ab + k0, lofs,                 BK, BM, (unsigned long long)ldA);
  tdm_load_2d(Bb + k0, lofs + A_ELEMS * 2u,  BK, BN, (unsigned long long)ldB);
#else
  (void)smem_base;
  // cooperative copy: 512 + 256 chunks of 16 B across 256 threads
#pragma unroll
  for (int c = tid; c < A_ELEMS / 8; c += 256) {
    int row = c >> 2, off = (c & 3) << 3;
    *(v8bf*)(sA + row * BK + off) = *(const v8bf*)(Ab + (size_t)row * ldA + k0 + off);
  }
  if (tid < B_ELEMS / 8) {
    int row = tid >> 2, off = (tid & 3) << 3;
    *(v8bf*)(sB + row * BK + off) = *(const v8bf*)(Bb + (size_t)row * ldB + k0 + off);
  }
#endif
}

// Generic NT GEMM: C[M,N] = A[M,K] (row-major) * Bt[N,K]^T.
__global__ __launch_bounds__(256) void k_gemm_nt(
    const bf16* __restrict__ A, const bf16* __restrict__ Bt,
    void* __restrict__ C,
    int Mrows, int Ncols, int K,
    int ldA, int ldB, int ldC,
    long long zA, long long zB, long long zC,
    float scale, int mode) {
  __shared__ bf16 smem[2 * BUF_ELEMS];   // 24 KB, double buffered

  const int tid  = threadIdx.x;
  const int lane = tid & 31;
  const int wid  = tid >> 5;
  const int wm   = (wid & 3) * 32;   // wave's M offset in block tile
  const int wn   = (wid >> 2) * 32;  // wave's N offset in block tile

  const int nb  = Ncols / BN;
  const int tmB = (blockIdx.x / nb) * BM;
  const int tnB = (blockIdx.x % nb) * BN;
  (void)Mrows;

  const bf16* Ab = A  + (size_t)(zA * blockIdx.z) + (size_t)tmB * ldA;
  const bf16* Bb = Bt + (size_t)(zB * blockIdx.z) + (size_t)tnB * ldB;

  const unsigned smem_base = (unsigned)(size_t)(void*)smem;  // LDS byte offset
  const int nk = K / BK;

  // Prologue: stage chunk 0 (and 1) -> bufs 0,1
  if (!USE_TDM || tid < 32) {
    stage_chunk(Ab, Bb, ldA, ldB, 0, smem, smem_base, 0, tid);
    if (nk > 1) stage_chunk(Ab, Bb, ldA, ldB, BK, smem, smem_base, 1, tid);
#if USE_TDM
    if (nk > 1) __builtin_amdgcn_s_wait_tensorcnt(2);
    else        __builtin_amdgcn_s_wait_tensorcnt(0);
#endif
  }
  __syncthreads();

  v8f a00 = {}, a01 = {}, a10 = {}, a11 = {};
  for (int kc = 0; kc < nk; ++kc) {
    const bf16* sA = smem + (kc & 1) * BUF_ELEMS;
    const bf16* sB = sA + A_ELEMS;
    v16bf fa0 = load_frag_a(sA + (size_t)wm * BK,        BK, 0, 0, lane);
    v16bf fa1 = load_frag_a(sA + (size_t)(wm + 16) * BK, BK, 0, 0, lane);
    v16bf fb0 = load_frag_b(sB + (size_t)wn * BK,        BK, 0, 0, lane);
    v16bf fb1 = load_frag_b(sB + (size_t)(wn + 16) * BK, BK, 0, 0, lane);
    a00 = __builtin_amdgcn_wmma_f32_16x16x32_bf16(false, fa0, false, fb0, (short)0, a00, false, false);
    a01 = __builtin_amdgcn_wmma_f32_16x16x32_bf16(false, fa0, false, fb1, (short)0, a01, false, false);
    a10 = __builtin_amdgcn_wmma_f32_16x16x32_bf16(false, fa1, false, fb0, (short)0, a10, false, false);
    a11 = __builtin_amdgcn_wmma_f32_16x16x32_bf16(false, fa1, false, fb1, (short)0, a11, false, false);

    __syncthreads();  // everyone done reading buf[kc&1]
    if (!USE_TDM || tid < 32) {
      if (kc + 2 < nk)
        stage_chunk(Ab, Bb, ldA, ldB, (kc + 2) * BK, smem, smem_base, kc & 1, tid);
#if USE_TDM
      if (kc + 2 < nk)      __builtin_amdgcn_s_wait_tensorcnt(2);
      else if (kc + 1 < nk) __builtin_amdgcn_s_wait_tensorcnt(0);
#endif
    }
    __syncthreads();  // buf[(kc+1)&1] ready for all waves
  }

  char* Cz = (char*)C + (size_t)(zC * blockIdx.z) * ((mode == 2) ? 4 : 2);
  store_tile(Cz, mode, ldC, tmB + wm,      tnB + wn,      lane, a00, scale);
  store_tile(Cz, mode, ldC, tmB + wm,      tnB + wn + 16, lane, a01, scale);
  store_tile(Cz, mode, ldC, tmB + wm + 16, tnB + wn,      lane, a10, scale);
  store_tile(Cz, mode, ldC, tmB + wm + 16, tnB + wn + 16, lane, a11, scale);
}

// x = bf16(emb[tok] * sqrt(D) + pos_enc)
__global__ void k_embed(const int* __restrict__ tok, const float* __restrict__ emb,
                        const float* __restrict__ pe, bf16* __restrict__ x) {
  size_t i = (size_t)blockIdx.x * blockDim.x + threadIdx.x;
  if (i >= (size_t)MTOK * DMODEL) return;
  int m = (int)(i / DMODEL);
  int d = (int)(i - (size_t)m * DMODEL);
  int l = m & (SEQ - 1);
  int t = tok[m];
  float v = emb[(size_t)t * DMODEL + d] * 32.0f + pe[(size_t)l * DMODEL + d];
  x[i] = (bf16)v;
}

// wT[e][d] = bf16(w[d][e])
__global__ void k_transposew(const float* __restrict__ w, bf16* __restrict__ wT) {
  int i = blockIdx.x * blockDim.x + threadIdx.x;
  if (i >= DMODEL * DMODEL) return;
  int e = i / DMODEL, d = i - e * DMODEL;
  wT[i] = (bf16)w[(size_t)d * DMODEL + e];
}

// Softmax over axis=1 (query axis): per (b,key) column stats over all queries.
__global__ void k_colstats(const bf16* __restrict__ qk, float* __restrict__ mx,
                           float* __restrict__ rden) {
  int idx = blockIdx.x * blockDim.x + threadIdx.x;
  if (idx >= BATCH * SEQ) return;
  int b = idx / SEQ, key = idx - b * SEQ;
  const bf16* p = qk + (size_t)b * SEQ * SEQ + key;
  float m = -3.0e38f;
  for (int q = 0; q < SEQ; ++q) { float v = (float)p[(size_t)q * SEQ]; m = fmaxf(m, v); }
  float s = 0.0f;
  for (int q = 0; q < SEQ; ++q) { float v = (float)p[(size_t)q * SEQ]; s += __expf(v - m); }
  mx[idx] = m;
  rden[idx] = 1.0f / s;
}

__global__ void k_softmax_apply(bf16* __restrict__ qk, const float* __restrict__ mx,
                                const float* __restrict__ rden) {
  size_t i = (size_t)blockIdx.x * blockDim.x + threadIdx.x;
  if (i >= (size_t)BATCH * SEQ * SEQ) return;
  int key = (int)(i & (SEQ - 1));
  int b   = (int)(i >> 22);  // SEQ*SEQ == 2^22
  int col = b * SEQ + key;
  float v = (float)qk[i];
  qk[i] = (bf16)(__expf(v - mx[col]) * rden[col]);
}

extern "C" void kernel_launch(void* const* d_in, const int* in_sizes, int n_in,
                              void* d_out, int out_size, void* d_ws, size_t ws_size,
                              hipStream_t stream) {
  (void)in_sizes; (void)n_in; (void)out_size; (void)ws_size;
  const int*   tok = (const int*)d_in[0];
  const float* emb = (const float*)d_in[1];
  const float* w_q = (const float*)d_in[2];
  const float* w_k = (const float*)d_in[3];
  const float* w_v = (const float*)d_in[4];
  const float* pe  = (const float*)d_in[5];

  char* ws = (char*)d_ws;
  size_t off = 0;
  auto carve = [&](size_t bytes) -> char* {
    char* p = ws + off;
    off += (bytes + 255) & ~(size_t)255;
    return p;
  };
  bf16*  x   = (bf16*)carve((size_t)MTOK * DMODEL * 2);       // 16 MB
  bf16*  wT  = (bf16*)carve((size_t)3 * DMODEL * DMODEL * 2); //  6 MB
  bf16*  q   = (bf16*)carve((size_t)MTOK * DMODEL * 2);       // 16 MB
  bf16*  k   = (bf16*)carve((size_t)MTOK * DMODEL * 2);       // 16 MB
  bf16*  vT  = (bf16*)carve((size_t)MTOK * DMODEL * 2);       // 16 MB, [D][MTOK]
  bf16*  qk  = (bf16*)carve((size_t)BATCH * SEQ * SEQ * 2);   // 32 MB
  float* mx  = (float*)carve((size_t)BATCH * SEQ * 4);
  float* rd  = (float*)carve((size_t)BATCH * SEQ * 4);

  const dim3 blk(256, 1, 1);  // 8 waves

  k_embed<<<(MTOK * DMODEL) / 256, blk, 0, stream>>>(tok, emb, pe, x);
  k_transposew<<<(DMODEL * DMODEL) / 256, blk, 0, stream>>>(w_q, wT);
  k_transposew<<<(DMODEL * DMODEL) / 256, blk, 0, stream>>>(w_k, wT + (size_t)DMODEL * DMODEL);
  k_transposew<<<(DMODEL * DMODEL) / 256, blk, 0, stream>>>(w_v, wT + (size_t)2 * DMODEL * DMODEL);

  // q = x@Wq, k = x@Wk (bf16 row-major); v stored transposed as vT[D][MTOK]
  const int gx_xw = (MTOK / BM) * (DMODEL / BN);  // 1024 blocks
  k_gemm_nt<<<dim3(gx_xw, 1, 1), blk, 0, stream>>>(
      x, wT, q, MTOK, DMODEL, DMODEL, DMODEL, DMODEL, DMODEL, 0, 0, 0, 1.0f, 0);
  k_gemm_nt<<<dim3(gx_xw, 1, 1), blk, 0, stream>>>(
      x, wT + (size_t)DMODEL * DMODEL, k, MTOK, DMODEL, DMODEL, DMODEL, DMODEL, DMODEL, 0, 0, 0, 1.0f, 0);
  k_gemm_nt<<<dim3(gx_xw, 1, 1), blk, 0, stream>>>(
      x, wT + (size_t)2 * DMODEL * DMODEL, vT, MTOK, DMODEL, DMODEL, DMODEL, DMODEL, MTOK, 0, 0, 0, 1.0f, 1);

  // qk[b] = (q_b @ k_b^T) / sqrt(D)
  const int gx_qk = (SEQ / BM) * (SEQ / BN);  // 512 blocks per batch
  k_gemm_nt<<<dim3(gx_qk, 1, BATCH), blk, 0, stream>>>(
      q, k, qk, SEQ, SEQ, DMODEL, DMODEL, DMODEL, SEQ,
      (long long)SEQ * DMODEL, (long long)SEQ * DMODEL, (long long)SEQ * SEQ,
      1.0f / 32.0f, 0);

  // softmax over the query axis (reference quirk), in place -> bf16 probs
  k_colstats<<<(BATCH * SEQ) / 256, blk, 0, stream>>>(qk, mx, rd);
  k_softmax_apply<<<(int)(((size_t)BATCH * SEQ * SEQ) / 256), blk, 0, stream>>>(qk, mx, rd);

  // out[b] = s_b @ v_b
  const int gx_sv = (SEQ / BM) * (DMODEL / BN);  // 256 blocks per batch
  k_gemm_nt<<<dim3(gx_sv, 1, BATCH), blk, 0, stream>>>(
      qk, vT, d_out, SEQ, DMODEL, SEQ, SEQ, MTOK, DMODEL,
      (long long)SEQ * SEQ, (long long)SEQ, (long long)SEQ * DMODEL,
      1.0f, 2);
}
